// GATLSTM_46600395161734
// MI455X (gfx1250) — matrix operand order repaired and checked by reading, not compiled, not run on hardware
//
#include <hip/hip_runtime.h>
#include <hip/hip_bf16.h>
#include <hip/hip_fp16.h>

typedef __attribute__((ext_vector_type(16))) _Float16 v16h;
typedef __attribute__((ext_vector_type(8)))  float    v8f;

// ---------- helpers ----------
__device__ __forceinline__ unsigned fenc(float x) {
    unsigned u = __float_as_uint(x);
    return (u & 0x80000000u) ? ~u : (u | 0x80000000u);
}
__device__ __forceinline__ float fdec(unsigned u) {
    return (u & 0x80000000u) ? __uint_as_float(u & 0x7fffffffu)
                             : __uint_as_float(~u);
}
__device__ __forceinline__ float sigf(float x) { return 1.f / (1.f + __expf(-x)); }

// ---------- prep: per-head attention scalars ----------
__global__ void k_prep_scal(const float* __restrict__ Wsrc, const float* __restrict__ Wdst,
                            const float* __restrict__ Wedge,
                            const float* __restrict__ att_src, const float* __restrict__ att_dst,
                            const float* __restrict__ att_edge, float* __restrict__ scal) {
    if (threadIdx.x == 0 && blockIdx.x == 0) {
        for (int h = 0; h < 2; h++) {
            float cs = 0.f, cd = 0.f, ce0 = 0.f, ce1 = 0.f;
            for (int o = 0; o < 32; o++) {
                int c = h * 32 + o;
                cs  += Wsrc[c] * att_src[c];
                cd  += Wdst[c] * att_dst[c];
                ce0 += Wedge[c * 2 + 0] * att_edge[c];
                ce1 += Wedge[c * 2 + 1] * att_edge[c];
            }
            scal[0 + h] = cs; scal[2 + h] = cd; scal[4 + h] = ce0; scal[6 + h] = ce1;
        }
    }
}

__global__ void k_prep_bias(const float* __restrict__ bih, const float* __restrict__ bhh,
                            float* __restrict__ b256) {
    int i = threadIdx.x;
    if (i < 256) b256[i] = bih[i] + bhh[i];
}

// ---------- prep: swizzle W (256x64, row-major) into WMMA B-fragment order ----------
// dst layout: [ntile(16)][kstep(2)][lane(32)][elem(16)]
// B is KxN (32x16 per fragment): lane -> n = ntile*16 + lane%16, group = lane/16,
//   elem e -> k = kstep*32 + (e<8 ? group*8+e : 16+group*8+(e-8)), value = W[n][k]
__global__ void k_swizzleB(const float* __restrict__ W, _Float16* __restrict__ dst) {
    int idx = blockIdx.x * blockDim.x + threadIdx.x;
    if (idx >= 16384) return;
    int e     = idx & 15;
    int lane  = (idx >> 4) & 31;
    int ks    = (idx >> 9) & 1;
    int ntile = idx >> 10;
    int j = ntile * 16 + (lane & 15);
    int k = ks * 32 + (lane >> 4) * 8 + (e & 7) + (e >> 3) * 16;
    dst[idx] = (_Float16)W[j * 64 + k];
}

// ---------- GAT: timestep-independent edge attention term ----------
__global__ void k_alpha(const float* __restrict__ ea, const float* __restrict__ scal,
                        float* __restrict__ ae, int E, int N) {
    int e = blockIdx.x * blockDim.x + threadIdx.x;
    if (e >= E + N) return;
    if (e < E) {
        float a0 = ea[e * 2 + 0], a1 = ea[e * 2 + 1];
        ae[e * 2 + 0] = a0 * scal[4] + a1 * scal[6];
        ae[e * 2 + 1] = a0 * scal[5] + a1 * scal[7];
    } else {
        ae[e * 2 + 0] = 0.f;
        ae[e * 2 + 1] = 0.f;
    }
}

__global__ void k_init(unsigned* __restrict__ menc, float* __restrict__ s,
                       float* __restrict__ tn, int n2) {
    int i = blockIdx.x * blockDim.x + threadIdx.x;
    if (i >= n2) return;
    menc[i] = 0x007FFFFFu;  // enc(-inf)
    s[i] = 0.f;
    tn[i] = 0.f;
}

__global__ void k_edge1(const float* __restrict__ x, const int* __restrict__ ei,
                        const float* __restrict__ ae, const float* __restrict__ scal,
                        float* __restrict__ z, unsigned* __restrict__ menc,
                        int E, int N, int t) {
    int e = blockIdx.x * blockDim.x + threadIdx.x;
    if (e >= E + N) return;
    int s, d;
    if (e < E) { s = ei[e]; d = ei[E + e]; } else { s = d = e - E; }
    float xs = x[s * 8 + t], xd = x[d * 8 + t];
#pragma unroll
    for (int h = 0; h < 2; h++) {
        float v = xs * scal[h] + xd * scal[2 + h] + ae[e * 2 + h];
        v = v > 0.f ? v : 0.2f * v;  // leaky_relu(0.2)
        z[e * 2 + h] = v;
        atomicMax(&menc[d * 2 + h], fenc(v));
    }
}

__global__ void k_edge2(const float* __restrict__ x, const int* __restrict__ ei,
                        const float* __restrict__ z, const unsigned* __restrict__ menc,
                        float* __restrict__ ssum, float* __restrict__ tnum,
                        int E, int N, int t) {
    int e = blockIdx.x * blockDim.x + threadIdx.x;
    if (e >= E + N) return;
    int s, d;
    if (e < E) { s = ei[e]; d = ei[E + e]; } else { s = d = e - E; }
    float xs = x[s * 8 + t];
#pragma unroll
    for (int h = 0; h < 2; h++) {
        float ex = __expf(z[e * 2 + h] - fdec(menc[d * 2 + h]));
        atomicAdd(&ssum[d * 2 + h], ex);
        atomicAdd(&tnum[d * 2 + h], ex * xs);
    }
}

// ---------- GAT epilogue + LayerNorm; one wave32 per node, 2 channels/lane ----------
__global__ void k_node(const float* __restrict__ tnum, const float* __restrict__ ssum,
                       const float* __restrict__ Wsrc, const float* __restrict__ gbias,
                       const float* __restrict__ gamma, const float* __restrict__ beta,
                       _Float16* __restrict__ hseq, int N, int t) {
    int n = blockIdx.x * 8 + (threadIdx.x >> 5);
    if (n >= N) return;
    int lane = threadIdx.x & 31;
    float t0 = tnum[n * 2 + 0] / (ssum[n * 2 + 0] + 1e-16f);
    float t1 = tnum[n * 2 + 1] / (ssum[n * 2 + 1] + 1e-16f);
    int c0 = lane, c1 = lane + 32;
    float p0 = t0 * Wsrc[c0] + gbias[c0];
    float p1 = t1 * Wsrc[c1] + gbias[c1];
    float sum = p0 + p1;
#pragma unroll
    for (int off = 16; off > 0; off >>= 1) sum += __shfl_xor(sum, off, 32);
    float mu = sum * (1.f / 64.f);
    float d0 = p0 - mu, d1 = p1 - mu;
    float vs = d0 * d0 + d1 * d1;
#pragma unroll
    for (int off = 16; off > 0; off >>= 1) vs += __shfl_xor(vs, off, 32);
    float rs = rsqrtf(vs * (1.f / 64.f) + 1e-5f);
    size_t base = ((size_t)t * N + n) * 64;
    hseq[base + c0] = (_Float16)(gamma[c0] * d0 * rs + beta[c0]);
    hseq[base + c1] = (_Float16)(gamma[c1] * d1 * rs + beta[c1]);
}

// ---------- A-fragment loader (ISA 7.12.2 16-bit A 16x32 layout) ----------
__device__ __forceinline__ v16h load_a_frag(const _Float16* Xs, int lane, int ks) {
    int m = lane & 15, grp = lane >> 4;
    const _Float16* p = Xs + m * 64 + ks * 32 + grp * 8;
    v16h a;
#pragma unroll
    for (int e = 0; e < 8; e++) { a[e] = p[e]; a[8 + e] = p[16 + e]; }
    return a;
}

// ---------- fused 8-step LSTM (WMMA) + predictor MLP; 16 nodes / block ----------
// Weight B-fragments are hoisted into registers (loop-invariant over the 8 steps):
// 16 fragments x 8 VGPRs = 128 VGPRs/lane, loaded once, fully coalesced from the
// pre-swizzled global image (L2-resident across all blocks).
__global__ __launch_bounds__(128)
void k_lstm(const _Float16* __restrict__ hseq, const _Float16* __restrict__ WbIh,
            const _Float16* __restrict__ WbHh, const float* __restrict__ b256,
            const float* __restrict__ Wp1, const float* __restrict__ bp1,
            const float* __restrict__ Wp2, const float* __restrict__ bp2,
            const float* __restrict__ x, float* __restrict__ out, int N) {
    __shared__ _Float16 sX[1024];      // 16 nodes x 64, f16 (A operand, X_t)
    __shared__ _Float16 sH[1024];      // hidden state, f16 (A operand)
    __shared__ float    sC[1024];      // cell state
    __shared__ float    sG[4096];      // gates 16 x 256
    __shared__ float    sHf[1024];     // final hidden, f32
    __shared__ float    sRed[128];

    int tid = threadIdx.x;
    int lane = tid & 31, wave = tid >> 5;
    int n0 = blockIdx.x * 16;

    // hoist weight B-fragments into registers (once)
    v16h bIh[4][2], bHh[4][2];
#pragma unroll
    for (int i = 0; i < 4; i++) {
        int ntile = wave * 4 + i;
#pragma unroll
        for (int ks = 0; ks < 2; ks++) {
            size_t off = ((size_t)(ntile * 2 + ks) * 32 + lane) * 16;
            bIh[i][ks] = *(const v16h*)(WbIh + off);
            bHh[i][ks] = *(const v16h*)(WbHh + off);
        }
    }

    for (int i = tid; i < 1024; i += 128) { sH[i] = (_Float16)0.f; sC[i] = 0.f; }
    __syncthreads();

    for (int t = 0; t < 8; t++) {
        // load X_t tile (16x64 f16 = 2 KB), one uint4 per thread
        {
            const uint4* src = (const uint4*)(hseq + ((size_t)t * N + n0) * 64);
            ((uint4*)sX)[tid] = src[tid];
        }
        __syncthreads();

        // each wave owns one gate (64 cols = 4 N-tiles): gates = X@Wih^T + H@Whh^T
#pragma unroll
        for (int i = 0; i < 4; i++) {
            int ntile = wave * 4 + i;
            v8f acc = {};
#pragma unroll
            for (int ks = 0; ks < 2; ks++) {
                v16h a = load_a_frag(sX, lane, ks);
                acc = __builtin_amdgcn_wmma_f32_16x16x32_f16(false, a, false, bIh[i][ks],
                                                             (short)0, acc, false, false);
            }
#pragma unroll
            for (int ks = 0; ks < 2; ks++) {
                v16h a = load_a_frag(sH, lane, ks);
                acc = __builtin_amdgcn_wmma_f32_16x16x32_f16(false, a, false, bHh[i][ks],
                                                             (short)0, acc, false, false);
            }
            int grp = lane >> 4, col = lane & 15;
#pragma unroll
            for (int r = 0; r < 8; r++) {
                int m = r + grp * 8;                     // C/D layout: VGPR r -> M
                sG[m * 256 + ntile * 16 + col] = acc[r];
            }
        }
        __syncthreads();

        // elementwise LSTM cell update (torch gate order i,f,g,o)
#pragma unroll
        for (int r = 0; r < 8; r++) {
            int id = tid + r * 128;
            int m = id >> 6, u = id & 63;
            float gi = sG[m * 256 + u]        + b256[u];
            float gf = sG[m * 256 + 64 + u]   + b256[64 + u];
            float gg = sG[m * 256 + 128 + u]  + b256[128 + u];
            float go = sG[m * 256 + 192 + u]  + b256[192 + u];
            float c = sC[id];
            c = sigf(gf) * c + sigf(gi) * tanhf(gg);
            float h = sigf(go) * tanhf(c);
            sC[id] = c;
            sH[id] = (_Float16)h;
            if (t == 7) sHf[id] = h;
        }
        __syncthreads();
    }

    // predictor: delta = relu(h@Wp1^T + bp1)@Wp2^T + bp2  (tiny, scalar)
    int m = tid >> 3, part = tid & 7;
    float sum = 0.f;
#pragma unroll
    for (int jj = 0; jj < 4; jj++) {
        int j = part * 4 + jj;
        float acc = bp1[j];
        for (int k = 0; k < 64; k++) acc += sHf[m * 64 + k] * Wp1[j * 64 + k];
        acc = acc > 0.f ? acc : 0.f;
        sum += acc * Wp2[j];
    }
    sRed[tid] = sum;
    __syncthreads();
    if (part == 0) {
        float tot = bp2[0];
        for (int q = 0; q < 8; q++) tot += sRed[m * 8 + q];
        out[n0 + m] = x[(size_t)(n0 + m) * 8 + 7] + tot;
    }
}

// ---------- host ----------
extern "C" void kernel_launch(void* const* d_in, const int* in_sizes, int n_in,
                              void* d_out, int out_size, void* d_ws, size_t ws_size,
                              hipStream_t stream) {
    const float* x        = (const float*)d_in[0];
    const int*   ei       = (const int*)d_in[1];
    const float* ea       = (const float*)d_in[2];
    const float* Wsrc     = (const float*)d_in[3];
    const float* Wdst     = (const float*)d_in[4];
    const float* Wedge    = (const float*)d_in[5];
    const float* att_src  = (const float*)d_in[6];
    const float* att_dst  = (const float*)d_in[7];
    const float* att_edge = (const float*)d_in[8];
    const float* gbias    = (const float*)d_in[9];
    const float* gamma    = (const float*)d_in[10];
    const float* beta     = (const float*)d_in[11];
    const float* Wih      = (const float*)d_in[12];
    const float* Whh      = (const float*)d_in[13];
    const float* bih      = (const float*)d_in[14];
    const float* bhh      = (const float*)d_in[15];
    const float* Wp1      = (const float*)d_in[16];
    const float* bp1      = (const float*)d_in[17];
    const float* Wp2      = (const float*)d_in[18];
    const float* bp2      = (const float*)d_in[19];
    float* out = (float*)d_out;

    int N  = in_sizes[0] / 8;   // T = 8
    int E  = in_sizes[1] / 2;
    int EP = E + N;

    char* w = (char*)d_ws;
    auto alloc = [&](size_t bytes) -> char* {
        char* p = w;
        w += (bytes + 255) & ~(size_t)255;
        return p;
    };
    float*     scal  = (float*)alloc(64);
    float*     b256  = (float*)alloc(1024);
    _Float16*  WbIh  = (_Float16*)alloc(32768);
    _Float16*  WbHh  = (_Float16*)alloc(32768);
    float*     ae    = (float*)alloc((size_t)EP * 2 * 4);
    float*     zb    = (float*)alloc((size_t)EP * 2 * 4);
    unsigned*  menc  = (unsigned*)alloc((size_t)N * 2 * 4);
    float*     ssum  = (float*)alloc((size_t)N * 2 * 4);
    float*     tnum  = (float*)alloc((size_t)N * 2 * 4);
    _Float16*  hseq  = (_Float16*)alloc((size_t)8 * N * 64 * 2);

    k_prep_scal<<<1, 1, 0, stream>>>(Wsrc, Wdst, Wedge, att_src, att_dst, att_edge, scal);
    k_prep_bias<<<1, 256, 0, stream>>>(bih, bhh, b256);
    k_swizzleB<<<64, 256, 0, stream>>>(Wih, WbIh);
    k_swizzleB<<<64, 256, 0, stream>>>(Whh, WbHh);

    int gE = (EP + 255) / 256;
    k_alpha<<<gE, 256, 0, stream>>>(ea, scal, ae, E, N);

    for (int t = 0; t < 8; t++) {
        k_init<<<(2 * N + 255) / 256, 256, 0, stream>>>(menc, ssum, tnum, 2 * N);
        k_edge1<<<gE, 256, 0, stream>>>(x, ei, ae, scal, zb, menc, E, N, t);
        k_edge2<<<gE, 256, 0, stream>>>(x, ei, zb, menc, ssum, tnum, E, N, t);
        k_node<<<(N + 7) / 8, 256, 0, stream>>>(tnum, ssum, Wsrc, gbias, gamma, beta, hseq, N, t);
    }

    k_lstm<<<N / 16, 128, 0, stream>>>(hseq, WbIh, WbHh, b256, Wp1, bp1, Wp2, bp2, x, out, N);
}
